// MultiHeadAttention_38070590111921
// MI455X (gfx1250) — matrix operand order repaired and checked by reading, not compiled, not run on hardware
//
#include <hip/hip_runtime.h>
#include <hip/hip_bf16.h>

#define HEADS 8
#define DH    64
#define DM    512
#define BATCH 4
#define SEQ   2048

typedef __attribute__((ext_vector_type(16))) __bf16 v16bf;
typedef __attribute__((ext_vector_type(8)))  float  v8f;
typedef __attribute__((ext_vector_type(4)))  unsigned int v4u;
typedef __attribute__((ext_vector_type(8)))  int    v8i;
typedef __attribute__((ext_vector_type(4)))  int    v4i;

// ---------------- WMMA fragment loaders (CDNA5 16x16x32 bf16 layouts) -------

// A operand: 16(M) x 32(K) bf16, row-major source with leading dim `ld`.
// lanes 0-15: row=lane, K 0..7 in elems 0..7, K 16..23 in elems 8..15
// lanes 16-31: row=lane-16, K 8..15 / 24..31
__device__ __forceinline__ v16bf load_fragA_bf16(const __bf16* base, int ld) {
  const int lane = threadIdx.x & 31;
  const int r = lane & 15, hi = lane >> 4;
  const __bf16* p = base + (size_t)r * ld + hi * 8;
  v16bf a;
#pragma unroll
  for (int j = 0; j < 8; ++j) a[j] = p[j];
#pragma unroll
  for (int j = 0; j < 8; ++j) a[j + 8] = p[16 + j];
  return a;
}

// Same as above but converting from fp32 source.
__device__ __forceinline__ v16bf load_fragA_f32(const float* base, int ld) {
  const int lane = threadIdx.x & 31;
  const int r = lane & 15, hi = lane >> 4;
  const float* p = base + (size_t)r * ld + hi * 8;
  v16bf a;
#pragma unroll
  for (int j = 0; j < 8; ++j) a[j] = (__bf16)p[j];
#pragma unroll
  for (int j = 0; j < 8; ++j) a[j + 8] = (__bf16)p[16 + j];
  return a;
}

// B operand: 32(K) x 16(N). Source is B^T, i.e. [N][K] row-major, lead dim ld.
// lane n holds column n; lanes 0-15: K 0..7 / 16..23, lanes 16-31: K 8..15 / 24..31.
__device__ __forceinline__ v16bf load_fragB_bf16(const __bf16* baseNK, int ld) {
  const int lane = threadIdx.x & 31;
  const int n = lane & 15, hi = lane >> 4;
  const __bf16* p = baseNK + (size_t)n * ld + hi * 8;
  v16bf b;
#pragma unroll
  for (int j = 0; j < 8; ++j) b[j] = p[j];
#pragma unroll
  for (int j = 0; j < 8; ++j) b[j + 8] = p[16 + j];
  return b;
}

__device__ __forceinline__ v8f wmma_bf16(v16bf a, v16bf b, v8f c) {
  return __builtin_amdgcn_wmma_f32_16x16x32_bf16(false, a, false, b, (short)0,
                                                 c, false, false);
}

// ---------------- TDM: async tile DMA, global (row-major) -> LDS ----------
// Loads a 64x64 bf16 tile from a [rows][64] bf16 tensor into LDS.
// Descriptor packing per CDNA5 ISA ch.8 (D# groups 0/1; groups 2/3 unused 2D).
// This toolchain exposes the 6-arg builtin:
//   (uint32x4 g0, int32x8 g1, int32x4 g2, int32x4 g3, int32x8 g4, i32 cpol)
__device__ __forceinline__ void tdm_load_tile64x64_bf16(const __bf16* gsrc,
                                                        unsigned lds_byte) {
  unsigned long long ga = (unsigned long long)(size_t)gsrc;
  v4u g0;
  g0[0] = 1u;                                   // count=1 (valid descriptor)
  g0[1] = lds_byte;                             // lds_addr
  g0[2] = (unsigned)(ga & 0xFFFFFFFFu);         // global_addr[31:0]
  g0[3] = (unsigned)((ga >> 32) & 0x01FFFFFFu)  // global_addr[56:32]
          | (2u << 30);                         // type=2 ("image")
  v8i g1;
  g1[0] = (int)(1u << 16);          // workgroup_mask=0, data_size=1 (2 bytes)
  g1[1] = (int)(64u << 16);         // tensor_dim0 = 64 (low 16 @ bit48)
  g1[2] = (int)(2048u << 16);       // tensor_dim1 = SEQ (low 16 @ bit80)
  g1[3] = (int)(64u << 16);         // tile_dim0 = 64 (@ bit112)
  g1[4] = 64;                       // tile_dim1 = 64, tile_dim2 = 0
  g1[5] = 64;                       // tensor_dim0_stride = 64 (low 32)
  g1[6] = 0;                        // stride0 hi, stride1 lo
  g1[7] = 0;                        // stride1 hi
  v4i gz4 = {0, 0, 0, 0};
  v8i gz8 = {0, 0, 0, 0, 0, 0, 0, 0};
  __builtin_amdgcn_tensor_load_to_lds(g0, g1, gz4, gz4, gz8, 0);
}

// ---------------- Kernel 1: weight prep (fp32 -> bf16, transpose to [N][K]) -

__global__ void mha_prep_weights(const float* __restrict__ Wq,
                                 const float* __restrict__ Wk,
                                 const float* __restrict__ Wv,
                                 const float* __restrict__ Wo,
                                 __bf16* __restrict__ WqT,
                                 __bf16* __restrict__ WkT,
                                 __bf16* __restrict__ WvT,
                                 __bf16* __restrict__ WoT) {
  int idx = blockIdx.x * 256 + threadIdx.x;
  if (idx < HEADS * DM * DH) {   // W[h][d][e] -> WT[h][e][d]
    int h = idx / (DM * DH);
    int r = idx - h * DM * DH;
    int d = r / DH, e = r - d * DH;
    size_t dst = ((size_t)h * DH + e) * DM + d;
    WqT[dst] = (__bf16)Wq[idx];
    WkT[dst] = (__bf16)Wk[idx];
    WvT[dst] = (__bf16)Wv[idx];
  }
  if (idx < DM * DM) {           // Wo[k][n] -> WoT[n][k]
    int k = idx / DM, n = idx - k * DM;
    WoT[(size_t)n * DM + k] = (__bf16)Wo[idx];
  }
}

// ---------------- Kernel 2: QKV projections -------------------------------
// grid: (SEQ/128, HEADS, BATCH*3). wave = 16 M rows x 64 N cols, K=512.

__global__ __launch_bounds__(256) void mha_qkv_proj(
    const float* __restrict__ query, const float* __restrict__ value,
    const __bf16* __restrict__ WqT, const __bf16* __restrict__ WkT,
    const __bf16* __restrict__ WvT, const float* __restrict__ bq,
    const float* __restrict__ bk, const float* __restrict__ bv,
    __bf16* __restrict__ qbuf, __bf16* __restrict__ kbuf,
    __bf16* __restrict__ vbuf) {
  const int mt = blockIdx.x;
  const int h = blockIdx.y;
  const int bz = blockIdx.z;
  const int b = bz / 3, which = bz % 3;

  const float* X = (which == 0 ? query : value) + (size_t)b * SEQ * DM;
  const __bf16* Wt = (which == 0 ? WqT : (which == 1 ? WkT : WvT)) +
                     (size_t)h * DH * DM;
  const float* bias = (which == 0 ? bq : (which == 1 ? bk : bv)) + h * DH;
  __bf16* outp = (which == 0 ? qbuf : (which == 1 ? kbuf : vbuf)) +
                 ((size_t)b * HEADS + h) * SEQ * DH;
  const float scale = (which == 0) ? 0.125f : 1.0f;  // fold 1/sqrt(64) into Q

  const int wave = threadIdx.x >> 5;
  const int m0 = mt * 128 + wave * 16;
  const int lane = threadIdx.x & 31;
  const int col = lane & 15, hi = lane >> 4;

  v8f acc[4];
#pragma unroll
  for (int c = 0; c < 4; ++c)
#pragma unroll
    for (int e = 0; e < 8; ++e) acc[c][e] = 0.f;

  for (int k0 = 0; k0 < DM; k0 += 32) {
    // Hoist all loads so the clause issues before the first wmma wait.
    v16bf a = load_fragA_f32(X + (size_t)m0 * DM + k0, DM);
    v16bf bw0 = load_fragB_bf16(Wt + (size_t)(0 * 16) * DM + k0, DM);
    v16bf bw1 = load_fragB_bf16(Wt + (size_t)(1 * 16) * DM + k0, DM);
    v16bf bw2 = load_fragB_bf16(Wt + (size_t)(2 * 16) * DM + k0, DM);
    v16bf bw3 = load_fragB_bf16(Wt + (size_t)(3 * 16) * DM + k0, DM);
    acc[0] = wmma_bf16(a, bw0, acc[0]);
    acc[1] = wmma_bf16(a, bw1, acc[1]);
    acc[2] = wmma_bf16(a, bw2, acc[2]);
    acc[3] = wmma_bf16(a, bw3, acc[3]);
  }
#pragma unroll
  for (int c = 0; c < 4; ++c) {
    int n = c * 16 + col;
#pragma unroll
    for (int e = 0; e < 8; ++e) {
      int row = hi * 8 + e;
      float v = (acc[c][e] + bias[n]) * scale;
      outp[(size_t)(m0 + row) * DH + n] = (__bf16)v;
    }
  }
}

// ---------------- Kernel 3: causal flash attention ------------------------
// grid: (SEQ/128, BATCH*HEADS). 8 waves/WG, wave = 16 q rows, key blocks of 64.
// K blocks stream in via TDM (double-buffered, TENSORcnt); V blocks go
// global->VGPR->LDS with an on-the-fly transpose for contiguous B-fragments.

__global__ __launch_bounds__(256) void mha_flash_attn(
    const __bf16* __restrict__ qbuf, const __bf16* __restrict__ kbuf,
    const __bf16* __restrict__ vbuf, __bf16* __restrict__ heads) {
  __shared__ __bf16 Ks[2][64 * 64];     // K blocks, row-major [t][e], 2 bufs
  __shared__ __bf16 Vt[64 * 64];        // V block transposed [e][t]
  __shared__ __bf16 Ps[8][16 * 64];     // per-wave P tile staging

  const int mt = blockIdx.x;
  const int bh = blockIdx.y;            // b*HEADS + h
  const int b = bh / HEADS, h = bh % HEADS;
  const __bf16* Q = qbuf + (size_t)bh * SEQ * DH;
  const __bf16* K = kbuf + (size_t)bh * SEQ * DH;
  const __bf16* V = vbuf + (size_t)bh * SEQ * DH;

  const int wave = threadIdx.x >> 5;
  const int lane = threadIdx.x & 31;
  const int col = lane & 15, hi = lane >> 4;
  const int qrow0 = mt * 128 + wave * 16;

  // Q fragments for e=0..31 and e=32..63, loaded once.
  v16bf aq0 = load_fragA_bf16(Q + (size_t)qrow0 * DH + 0, DH);
  v16bf aq1 = load_fragA_bf16(Q + (size_t)qrow0 * DH + 32, DH);

  v8f o[4];
  float m_i[8], l_i[8];
#pragma unroll
  for (int c = 0; c < 4; ++c)
#pragma unroll
    for (int e = 0; e < 8; ++e) o[c][e] = 0.f;
#pragma unroll
  for (int e = 0; e < 8; ++e) { m_i[e] = -3.0e38f; l_i[e] = 0.f; }

  const int kmax = mt * 128 + 128;      // causal upper bound for this WG
  int cur = 0;
  if (wave == 0)                        // prime the pipeline: TDM block 0
    tdm_load_tile64x64_bf16(K, (unsigned)(size_t)&Ks[0][0]);

  for (int kb0 = 0; kb0 < kmax; kb0 += 64) {
    // V block: global -> registers -> transposed LDS store (overlaps TDM).
    {
      int t = threadIdx.x;
      int row = t >> 2, seg = (t & 3) * 16;
      const __bf16* gv = V + (size_t)(kb0 + row) * DH + seg;
#pragma unroll
      for (int j = 0; j < 16; ++j) Vt[(seg + j) * 64 + row] = gv[j];
    }
    if (wave == 0) __builtin_amdgcn_s_wait_tensorcnt(0);  // K block landed
    __syncthreads();
    // Kick next K block into the other buffer while we compute this one.
    if (wave == 0 && kb0 + 64 < kmax)
      tdm_load_tile64x64_bf16(K + (size_t)(kb0 + 64) * DH,
                              (unsigned)(size_t)&Ks[cur ^ 1][0]);

    if (kb0 <= qrow0 + 15) {            // wave has visible keys in this block
      const __bf16* Kc = &Ks[cur][0];
      // ---- scores S = Q K^T (scale pre-folded into Q) ----
      v8f s[4];
#pragma unroll
      for (int c = 0; c < 4; ++c) {
        v8f z;
#pragma unroll
        for (int e = 0; e < 8; ++e) z[e] = 0.f;
        v16bf bk0 = load_fragB_bf16(Kc + (c * 16) * 64 + 0, 64);
        v16bf bk1 = load_fragB_bf16(Kc + (c * 16) * 64 + 32, 64);
        z = wmma_bf16(aq0, bk0, z);
        z = wmma_bf16(aq1, bk1, z);
        s[c] = z;
      }
      // ---- causal mask: t > q  ->  -1e9 (matches reference semantics) ----
#pragma unroll
      for (int c = 0; c < 4; ++c) {
        int t_idx = kb0 + c * 16 + col;
#pragma unroll
        for (int e = 0; e < 8; ++e) {
          int q_idx = qrow0 + hi * 8 + e;
          if (t_idx > q_idx) s[c][e] = -1.0e9f;
        }
      }
      // ---- online softmax update (row stats live in a 16-lane half) ----
#pragma unroll
      for (int e = 0; e < 8; ++e) {
        float mb = fmaxf(fmaxf(s[0][e], s[1][e]), fmaxf(s[2][e], s[3][e]));
#pragma unroll
        for (int off = 1; off < 16; off <<= 1)
          mb = fmaxf(mb, __shfl_xor(mb, off, 32));
        float mn = fmaxf(m_i[e], mb);
        float alpha = __expf(m_i[e] - mn);
        m_i[e] = mn;
        l_i[e] *= alpha;
#pragma unroll
        for (int c = 0; c < 4; ++c) o[c][e] *= alpha;
      }
      // ---- P = exp(S - m); stage to LDS for layout transpose ----
      __bf16* Pw = &Ps[wave][0];
      float psum[8];
#pragma unroll
      for (int e = 0; e < 8; ++e) psum[e] = 0.f;
#pragma unroll
      for (int c = 0; c < 4; ++c)
#pragma unroll
        for (int e = 0; e < 8; ++e) {
          float p = __expf(s[c][e] - m_i[e]);
          psum[e] += p;
          Pw[(hi * 8 + e) * 64 + c * 16 + col] = (__bf16)p;
        }
#pragma unroll
      for (int e = 0; e < 8; ++e) {
        float v = psum[e];
#pragma unroll
        for (int off = 1; off < 16; off <<= 1) v += __shfl_xor(v, off, 32);
        l_i[e] += v;
      }
      // ---- O += P V ----
      v16bf ap0 = load_fragA_bf16(Pw + 0, 64);
      v16bf ap1 = load_fragA_bf16(Pw + 32, 64);
#pragma unroll
      for (int ce = 0; ce < 4; ++ce) {
        v16bf bv0 = load_fragB_bf16(&Vt[(ce * 16) * 64 + 0], 64);
        v16bf bv1 = load_fragB_bf16(&Vt[(ce * 16) * 64 + 32], 64);
        o[ce] = wmma_bf16(ap0, bv0, o[ce]);
        o[ce] = wmma_bf16(ap1, bv1, o[ce]);
      }
    }
    __syncthreads();
    cur ^= 1;
  }

  // ---- normalize and emit heads[b][s][h*64 + e] as bf16 ----
#pragma unroll
  for (int ce = 0; ce < 4; ++ce) {
#pragma unroll
    for (int e = 0; e < 8; ++e) {
      int row = qrow0 + hi * 8 + e;
      float v = o[ce][e] / l_i[e];
      heads[((size_t)b * SEQ + row) * DM + h * DH + ce * 16 + col] = (__bf16)v;
    }
  }
}

// ---------------- Kernel 4: output projection -----------------------------
// grid: (B*S/128, DM/64). wave = 16 x 64 tile, K = 512, fp32 out + bias.

__global__ __launch_bounds__(256) void mha_out_proj(
    const __bf16* __restrict__ heads, const __bf16* __restrict__ WoT,
    const float* __restrict__ bo, float* __restrict__ out) {
  const int mt = blockIdx.x;
  const int nt = blockIdx.y;
  const int wave = threadIdx.x >> 5;
  const int lane = threadIdx.x & 31;
  const int col = lane & 15, hi = lane >> 4;
  const int m0 = mt * 128 + wave * 16;
  const int n0 = nt * 64;

  v8f acc[4];
#pragma unroll
  for (int c = 0; c < 4; ++c)
#pragma unroll
    for (int e = 0; e < 8; ++e) acc[c][e] = 0.f;

  for (int k0 = 0; k0 < DM; k0 += 32) {
    v16bf a = load_fragA_bf16(heads + (size_t)m0 * DM + k0, DM);
    v16bf bw0 = load_fragB_bf16(WoT + (size_t)(n0 + 0 * 16) * DM + k0, DM);
    v16bf bw1 = load_fragB_bf16(WoT + (size_t)(n0 + 1 * 16) * DM + k0, DM);
    v16bf bw2 = load_fragB_bf16(WoT + (size_t)(n0 + 2 * 16) * DM + k0, DM);
    v16bf bw3 = load_fragB_bf16(WoT + (size_t)(n0 + 3 * 16) * DM + k0, DM);
    acc[0] = wmma_bf16(a, bw0, acc[0]);
    acc[1] = wmma_bf16(a, bw1, acc[1]);
    acc[2] = wmma_bf16(a, bw2, acc[2]);
    acc[3] = wmma_bf16(a, bw3, acc[3]);
  }
#pragma unroll
  for (int c = 0; c < 4; ++c) {
    int n = n0 + c * 16 + col;
#pragma unroll
    for (int e = 0; e < 8; ++e) {
      int row = m0 + hi * 8 + e;
      out[(size_t)row * DM + n] = acc[c][e] + bo[n];
    }
  }
}

// ---------------- host launcher -------------------------------------------

extern "C" void kernel_launch(void* const* d_in, const int* in_sizes, int n_in,
                              void* d_out, int out_size, void* d_ws,
                              size_t ws_size, hipStream_t stream) {
  const float* query = (const float*)d_in[0];
  const float* value = (const float*)d_in[1];
  // d_in[2] = causal mask: handled analytically, never read (saves 16 MB).
  const float* Wq = (const float*)d_in[3];
  const float* bq = (const float*)d_in[4];
  const float* Wk = (const float*)d_in[5];
  const float* bk = (const float*)d_in[6];
  const float* Wv = (const float*)d_in[7];
  const float* bv = (const float*)d_in[8];
  const float* Wo = (const float*)d_in[9];
  const float* bo = (const float*)d_in[10];
  float* out = (float*)d_out;

  char* ws = (char*)d_ws;
  size_t off = 0;
  auto walloc = [&](size_t bytes) -> void* {
    void* p = ws + off;
    off += (bytes + 255) & ~(size_t)255;
    return p;
  };
  __bf16* WqT = (__bf16*)walloc((size_t)HEADS * DM * DH * 2);
  __bf16* WkT = (__bf16*)walloc((size_t)HEADS * DM * DH * 2);
  __bf16* WvT = (__bf16*)walloc((size_t)HEADS * DM * DH * 2);
  __bf16* WoT = (__bf16*)walloc((size_t)DM * DM * 2);
  __bf16* qbuf = (__bf16*)walloc((size_t)BATCH * HEADS * SEQ * DH * 2);
  __bf16* kbuf = (__bf16*)walloc((size_t)BATCH * HEADS * SEQ * DH * 2);
  __bf16* vbuf = (__bf16*)walloc((size_t)BATCH * HEADS * SEQ * DH * 2);
  __bf16* heads = (__bf16*)walloc((size_t)BATCH * SEQ * DM * 2);

  mha_prep_weights<<<(HEADS * DM * DH + 255) / 256, 256, 0, stream>>>(
      Wq, Wk, Wv, Wo, WqT, WkT, WvT, WoT);

  dim3 g1(SEQ / 128, HEADS, BATCH * 3);
  mha_qkv_proj<<<g1, 256, 0, stream>>>(query, value, WqT, WkT, WvT, bq, bk, bv,
                                       qbuf, kbuf, vbuf);

  dim3 g2(SEQ / 128, BATCH * HEADS);
  mha_flash_attn<<<g2, 256, 0, stream>>>(qbuf, kbuf, vbuf, heads);

  dim3 g3((BATCH * SEQ) / 128, DM / 64);
  mha_out_proj<<<g3, 256, 0, stream>>>(heads, WoT, bo, out);
}
